// TTNetSerializedWithState_78013785964662
// MI455X (gfx1250) — matrix operand order repaired and checked by reading, not compiled, not run on hardware
//
#include <hip/hip_runtime.h>
#include <stdint.h>

// ---- problem dims (match reference) ----
#define D_STEPS 16
#define BSZ     256
#define MDIM    192
#define RDIM    192

// ---- tiling ----
#define BM   64                 // batch rows per workgroup
#define BN   96                 // output cols per workgroup
#define IC   8                  // split-K chunk over rank index i
#define NKST (IC * (MDIM / 32)) // 48 K-steps of 32 per WG

// TDM lands x with pad_interval=64 DW, pad_amount=4 DW:
// float offset of x[row][m] in LDS = row*204 + m + 4*(m/64)
#define XSS  204                // padded x row stride (floats)
#define SSS  9                  // LDS s row stride
#define WTS  40                 // LDS W^T row stride in bf16 (32 used, padded for b128 align/banks)

typedef __attribute__((ext_vector_type(16))) __bf16 v16bf;
typedef __attribute__((ext_vector_type(8)))  __bf16 v8bf;
typedef __attribute__((ext_vector_type(8)))  float  v8f;
typedef __attribute__((ext_vector_type(4)))  uint32_t u32x4;
typedef __attribute__((ext_vector_type(8)))  int      i32x8;
typedef __attribute__((ext_vector_type(4)))  int      i32x4;

__global__ __launch_bounds__(256) void tt_zero(float* __restrict__ p) {
  ((float4*)p)[blockIdx.x * 256 + threadIdx.x] = make_float4(0.f, 0.f, 0.f, 0.f);
}

// One scan step:  s_out[b,j] += sum_{i in chunk, m} s_in[b,i] * W[i,m,j] * x[b,m]
__global__ __launch_bounds__(256) void tt_step(const float* __restrict__ x,    // [BSZ, MDIM]
                                               const float* __restrict__ s_in, // [BSZ, RDIM]
                                               const float* __restrict__ W,    // [RDIM, MDIM, RDIM]
                                               float* __restrict__ s_out)      // [BSZ, RDIM], pre-zeroed
{
  __shared__ float  xs[BM * XSS];      // x tile, TDM-landed with padding
  __shared__ float  ss[BM * SSS];
  __shared__ __bf16 wt[2][96 * WTS];   // double-buffered 32x96 W tile, transposed, bf16

  const int tid  = threadIdx.x;
  const int lane = tid & 31;
  const int wave = tid >> 5;
  const int wr   = wave >> 1;          // row tile 0..3 (16 rows each)
  const int wc   = wave & 1;           // col half 0..1 (48 cols each)

  const int b0 = blockIdx.x * BM;
  const int j0 = blockIdx.y * BN;
  const int i0 = blockIdx.z * IC;

  // ---- x tile [BM x MDIM] via Tensor Data Mover (wave 0 issues, TENSORcnt tracks) ----
  if (tid < 32) {
    const uint64_t ga  = (uint64_t)(uintptr_t)(x + (size_t)b0 * MDIM);
    const uint32_t lds = (uint32_t)(uintptr_t)(&xs[0]);
    // D# group0: count=1, lds_addr, 57b global_addr, type=2
    u32x4 g0 = { 1u, lds, (uint32_t)ga,
                 (uint32_t)((ga >> 32) & 0x01FFFFFFu) | (2u << 30) };
    // D# group1: data_size=4B(2), pad_enable, pad_interval=64DW(5), pad_amount=4DW(3),
    //            tensor_dim0=192, tensor_dim1=64, tile_dim0=192, tile_dim1=64, stride0=192
    const uint32_t td0 = MDIM, td1 = BM, t0 = MDIM, t1 = BM;
    const uint64_t st0 = MDIM;
    i32x8 g1 = { (int)((2u << 16) | (1u << 20) | (5u << 22) | (3u << 25)),
                 (int)((td0 & 0xFFFFu) << 16),
                 (int)((td0 >> 16) | ((td1 & 0xFFFFu) << 16)),
                 (int)((td1 >> 16) | (t0 << 16)),
                 (int)(t1),                       // tile_dim1 | tile_dim2(0)<<16
                 (int)(uint32_t)st0,
                 (int)(uint32_t)(st0 >> 32),      // stride0.hi | stride1.lo(0)
                 0 };
    i32x4 gz = { 0, 0, 0, 0 };
#if __has_include(<hip/amd_detail/amd_gfx1250_TDM.h>)
    i32x8 gz8 = { 0, 0, 0, 0, 0, 0, 0, 0 };
    __builtin_amdgcn_tensor_load_to_lds(g0, g1, gz, gz, gz8, 0);   // 6-arg toolchain
#else
    __builtin_amdgcn_tensor_load_to_lds(g0, g1, gz, gz, 0);        // ROCm 7.2 (5-arg)
#endif
    __builtin_amdgcn_s_wait_tensorcnt(0);
  }
  // ---- s tile [BM x IC] ----
  #pragma unroll
  for (int r = 0; r < 2; ++r) {
    int e = tid + 256 * r;                   // 0..511
    int row = e / IC, ii = e % IC;
    ss[row * SSS + ii] = s_in[(size_t)(b0 + row) * RDIM + i0 + ii];
  }

  // ---- loop-invariant offsets (hoisted: no div/mod in the hot loop) ----
  int soff[6], toff[6];
  #pragma unroll
  for (int r = 0; r < 6; ++r) {
    const int p  = tid + 256 * r;            // 0..1535 k-pairs
    const int kp = p / 96, c = p % 96;       // k = 2*kp
    soff[r] = 2 * kp * RDIM + c;             // f32 offset into current W tile
    toff[r] = c * WTS + 2 * kp;              // bf16 offset into wt (4B-aligned store)
  }
  const int pfoff = (tid / 3) * RDIM + (tid % 3) * 32;   // prefetch lane offset (96 lanes)

  // W tile pointer is LINEAR in ks: tile(ks) = W + (i0*MDIM + 32*ks)*RDIM + j0
  const float* Wk = W + ((size_t)i0 * MDIM) * RDIM + j0;

  auto stage = [&](int buf, const float* Wb) {
    __bf16* wb = &wt[buf][0];
    #pragma unroll
    for (int r = 0; r < 6; ++r) {
      float f0 = Wb[soff[r]];
      float f1 = Wb[soff[r] + RDIM];
      union { __bf16 h[2]; uint32_t u; } cv;
      cv.h[0] = (__bf16)f0; cv.h[1] = (__bf16)f1;
      *(uint32_t*)(wb + toff[r]) = cv.u;
    }
  };

  stage(0, Wk);
  __syncthreads();

  v8f acc0 = {}, acc1 = {}, acc2 = {};

  const int koff  = (lane < 16) ? 0 : 8;     // A-frag K offset (ISA 16-bit A layout)
  const int K0    = (lane < 16) ? 0 : 8;     // B-frag K offset (same table order as A)
  const int rw    = wr * 16 + (lane & 15);   // my row within the b-tile
  const int rbase = rw * XSS + koff;

  const float* Wst = Wk + 32 * RDIM;         // stage-ahead pointer (ks+1)
  const float* Wpf = Wk + 64 * RDIM;         // prefetch pointer   (ks+2)

  int mm = 0, il = 0;                        // m0 = mm*32 ; i-index within chunk
  for (int ks = 0; ks < NKST; ++ks) {
    // prefetch W tile two K-steps ahead: 96 lanes x 128B lines cover 32x96 f32
    if (ks + 2 < NKST && tid < 96) __builtin_prefetch(Wpf + pfoff, 0, 1);
    if (ks + 1 < NKST) stage((ks & 1) ^ 1, Wst);   // overlap next tile stage

    // A fragment: T[b,k] = s[b,i] * x[b,m], built in f32, converted once to bf16
    const float sv   = ss[rw * SSS + il];
    const int   base = rbase + mm * 32 + 4 * (mm >> 1);  // TDM pad-aware x offset
    const float4* xp = (const float4*)(xs + base);
    const float4 xa = xp[0], xb = xp[1];                 // K = koff .. koff+7
    const float4* xq = (const float4*)(xs + base + 16);
    const float4 xc = xq[0], xd = xq[1];                 // K = 16+koff .. 16+koff+7
    v16bf a;
    a[0]=(__bf16)(sv*xa.x); a[1]=(__bf16)(sv*xa.y); a[2] =(__bf16)(sv*xa.z); a[3] =(__bf16)(sv*xa.w);
    a[4]=(__bf16)(sv*xb.x); a[5]=(__bf16)(sv*xb.y); a[6] =(__bf16)(sv*xb.z); a[7] =(__bf16)(sv*xb.w);
    a[8]=(__bf16)(sv*xc.x); a[9]=(__bf16)(sv*xc.y); a[10]=(__bf16)(sv*xc.z); a[11]=(__bf16)(sv*xc.w);
    a[12]=(__bf16)(sv*xd.x); a[13]=(__bf16)(sv*xd.y); a[14]=(__bf16)(sv*xd.z); a[15]=(__bf16)(sv*xd.w);

    const __bf16* wb = &wt[ks & 1][0];
    #pragma unroll
    for (int nt = 0; nt < 3; ++nt) {
      const int col = wc * 48 + nt * 16 + (lane & 15);
      const v8bf* bp = (const v8bf*)(wb + col * WTS + K0);   // 16B aligned
      v8bf blo = bp[0], bhi = bp[2];
      v16bf b = __builtin_shufflevector(blo, bhi, 0,1,2,3,4,5,6,7,8,9,10,11,12,13,14,15);
      v8f* ap = (nt == 0) ? &acc0 : (nt == 1) ? &acc1 : &acc2;
      *ap = __builtin_amdgcn_wmma_f32_16x16x32_bf16(false, a, false, b, (short)0, *ap, false, false);
    }
    __syncthreads();

    Wst += 32 * RDIM;
    Wpf += 32 * RDIM;
    if (++mm == 6) { mm = 0; ++il; }
  }

  // split-K reduction: atomic f32 accumulate into pre-zeroed state
  const int rowhi = (lane < 16) ? 0 : 8;           // C/D layout: VGPR r -> M = r + 8*(lane>=16)
  #pragma unroll
  for (int nt = 0; nt < 3; ++nt) {
    const v8f& av = (nt == 0) ? acc0 : (nt == 1) ? acc1 : acc2;
    const int col = j0 + wc * 48 + nt * 16 + (lane & 15);
    #pragma unroll
    for (int r = 0; r < 8; ++r) {
      float* p = s_out + (size_t)(b0 + wr * 16 + r + rowhi) * RDIM + col;
      __hip_atomic_fetch_add(p, av[r], __ATOMIC_RELAXED, __HIP_MEMORY_SCOPE_AGENT);
    }
  }
}

// log_softmax over rows of [BSZ, RDIM]; one wave per row
__global__ __launch_bounds__(256) void tt_lsm(const float* __restrict__ in, float* __restrict__ out) {
  const int lane = threadIdx.x & 31, wave = threadIdx.x >> 5;
  const int row  = blockIdx.x * 8 + wave;
  const float* r = in + (size_t)row * RDIM;
  float v[6], mx = -3.402823e38f;
  #pragma unroll
  for (int t = 0; t < 6; ++t) { v[t] = r[lane + 32 * t]; mx = fmaxf(mx, v[t]); }
  #pragma unroll
  for (int m = 16; m; m >>= 1) mx = fmaxf(mx, __shfl_xor(mx, m, 32));
  float s = 0.f;
  #pragma unroll
  for (int t = 0; t < 6; ++t) s += expf(v[t] - mx);
  #pragma unroll
  for (int m = 16; m; m >>= 1) s += __shfl_xor(s, m, 32);
  const float l = logf(s) + mx;
  float* o = out + (size_t)row * RDIM;
  #pragma unroll
  for (int t = 0; t < 6; ++t) o[lane + 32 * t] = v[t] - l;
}

extern "C" void kernel_launch(void* const* d_in, const int* in_sizes, int n_in,
                              void* d_out, int out_size, void* d_ws, size_t ws_size,
                              hipStream_t stream) {
  (void)in_sizes; (void)n_in; (void)out_size; (void)ws_size;
  const float* tensor = (const float*)d_in[0];   // [D, B, M]
  const float* state0 = (const float*)d_in[1];   // [B, R]
  const float* Ws     = (const float*)d_in[2];   // [D, R, M, R]
  float* out = (float*)d_out;

  float* sb0 = (float*)d_ws;                     // ping-pong state: 2 * 256*192 f32 = 384 KB
  float* sb1 = sb0 + BSZ * RDIM;

  const dim3 grid(BSZ / BM, RDIM / BN, RDIM / IC);   // 4 x 2 x 24 = 192 WGs/step
  for (int d = 0; d < D_STEPS; ++d) {
    const float* sin = (d == 0) ? state0 : ((d & 1) ? sb0 : sb1);
    float* sout = (d & 1) ? sb1 : sb0;
    tt_zero<<<BSZ * RDIM / 4 / 256, 256, 0, stream>>>(sout);
    tt_step<<<grid, 256, 0, stream>>>(tensor + (size_t)d * BSZ * MDIM, sin,
                                      Ws + (size_t)d * RDIM * MDIM * RDIM, sout);
  }
  tt_lsm<<<BSZ / 8, 256, 0, stream>>>(((D_STEPS - 1) & 1) ? sb1 : sb0, out);
}
